// PQcodebook_50337016709477
// MI455X (gfx1250) — compile-verified, matrix operand order checked by reading, not run on hardware
//
#include <hip/hip_runtime.h>

typedef _Float16 half_t;
typedef __attribute__((ext_vector_type(16))) _Float16 v16h;
typedef __attribute__((ext_vector_type(8)))  _Float16 v8h;
typedef __attribute__((ext_vector_type(8)))  float    v8f;
typedef __attribute__((ext_vector_type(4)))  float    v4f;

#define B_N    16384
#define M_SUB  8
#define K_CB   1024
#define D_FULL 512
#define D_SUB  64

// ---------------------------------------------------------------------------
// Kernel 1: codebook fp32 -> fp16 (for WMMA) + e2 = ||c||^2.
// 8192 rows (m,k), one thread per row of 64 floats.
// ---------------------------------------------------------------------------
__global__ __launch_bounds__(256)
void cvt_cb_e2_kernel(const float* __restrict__ cb,
                      half_t* __restrict__ cbh,
                      float* __restrict__ e2) {
    const int idx = blockIdx.x * 256 + threadIdx.x;           // 0..8191
    const float* src = cb + (size_t)idx * D_SUB;
    half_t* dst = cbh + (size_t)idx * D_SUB;
    float s = 0.0f;
#pragma unroll
    for (int j = 0; j < D_SUB; j += 4) {
        const v4f v = *(const v4f*)(src + j);
        s += v.x * v.x + v.y * v.y + v.z * v.z + v.w * v.w;
        dst[j + 0] = (half_t)v.x;
        dst[j + 1] = (half_t)v.y;
        dst[j + 2] = (half_t)v.z;
        dst[j + 3] = (half_t)v.w;
    }
    e2[idx] = s;
}

// ---------------------------------------------------------------------------
// Kernel 2: WMMA distance + argmin.
// Grid: (B/256, M) blocks x 256 threads (8 waves). Each wave owns TWO 16-row
// b-tiles (32 rows) for one subspace m and scans K=1024 in 16-col tiles.
// Per k-tile: 4x v_wmma_f32_16x16x32_f16 sharing one set of B operands,
// software-pipelined (double-buffered) B / e2 loads; loop unrolled x2 so the
// two buffers alternate via register renaming (no v_mov rotation).
// ---------------------------------------------------------------------------
__global__ __launch_bounds__(256)
void pq_argmin_kernel(const float* __restrict__ z,
                      const half_t* __restrict__ cbh,
                      const float* __restrict__ e2,
                      int* __restrict__ kout) {
    const int tid  = threadIdx.x;
    const int wave = tid >> 5;
    const int lane = tid & 31;
    const int hf   = lane >> 4;       // wave half: selects K(+8) / row(+8)
    const int l16  = lane & 15;
    const int m    = blockIdx.y;
    const int btile = (blockIdx.x * 8 + wave) * 32;
    const int b0o  = hf * 8;          // per-lane d-offset base

    // ---- A operands: two z row-tiles, fp32 -> fp16, loaded once ----
    // 16-bit A 16x32 layout: element e -> K = (e<8 ? e : e+8) + hf*8.
    v16h A0a, A1a, A0b, A1b;
    {
        const float* zr0 = z + (size_t)(btile + l16) * D_FULL + m * D_SUB;
        const float* zr1 = zr0 + 16 * D_FULL;
#pragma unroll
        for (int j = 0; j < 8; ++j) {
            A0a[j]     = (half_t)zr0[b0o + j];
            A0a[j + 8] = (half_t)zr0[b0o + 16 + j];
            A1a[j]     = (half_t)zr0[b0o + 32 + j];
            A1a[j + 8] = (half_t)zr0[b0o + 48 + j];
            A0b[j]     = (half_t)zr1[b0o + j];
            A0b[j + 8] = (half_t)zr1[b0o + 16 + j];
            A1b[j]     = (half_t)zr1[b0o + 32 + j];
            A1b[j + 8] = (half_t)zr1[b0o + 48 + j];
        }
    }

    float bestv[16];
    int   bestk[16];
#pragma unroll
    for (int i = 0; i < 16; ++i) { bestv[i] = 3.4e38f; bestk[i] = 0; }

    const half_t* cbm = cbh + (size_t)m * K_CB * D_SUB;
    const float*  e2m = e2 + m * K_CB;

    // ---- prologue: loads for k-tile 0 ----
    const half_t* crow = cbm + (size_t)l16 * D_SUB;
    v8h nb0a = *(const v8h*)(crow + b0o);
    v8h nb0b = *(const v8h*)(crow + b0o + 16);
    v8h nb1a = *(const v8h*)(crow + b0o + 32);
    v8h nb1b = *(const v8h*)(crow + b0o + 48);
    float ne2 = e2m[l16];

#pragma unroll 2
    for (int kt = 0; kt < K_CB / 16; ++kt) {
        const int kcol = kt * 16 + l16;                 // this lane's column
        const v8h b0a = nb0a, b0b = nb0b, b1a = nb1a, b1b = nb1b;
        const float e2v = ne2;

        // ---- issue next k-tile's loads early (hidden under WMMA+epilogue) ----
        if (kt + 1 < K_CB / 16) {
            const half_t* nrow = cbm + (size_t)(kcol + 16) * D_SUB;
            __builtin_prefetch(nrow + 16 * D_SUB, 0, 1);   // global_prefetch_b8
            nb0a = *(const v8h*)(nrow + b0o);
            nb0b = *(const v8h*)(nrow + b0o + 16);
            nb1a = *(const v8h*)(nrow + b0o + 32);
            nb1b = *(const v8h*)(nrow + b0o + 48);
            ne2  = e2m[kcol + 16];
        }

        const v16h B0 = __builtin_shufflevector(b0a, b0b,
            0,1,2,3,4,5,6,7,8,9,10,11,12,13,14,15);
        const v16h B1 = __builtin_shufflevector(b1a, b1b,
            0,1,2,3,4,5,6,7,8,9,10,11,12,13,14,15);

        v8f c0 = {0.f,0.f,0.f,0.f,0.f,0.f,0.f,0.f};
        v8f c1 = {0.f,0.f,0.f,0.f,0.f,0.f,0.f,0.f};
        // interleave the two independent accumulator chains
        c0 = __builtin_amdgcn_wmma_f32_16x16x32_f16(false, A0a, false, B0,
                                                    (short)0, c0, false, false);
        c1 = __builtin_amdgcn_wmma_f32_16x16x32_f16(false, A0b, false, B0,
                                                    (short)0, c1, false, false);
        c0 = __builtin_amdgcn_wmma_f32_16x16x32_f16(false, A1a, false, B1,
                                                    (short)0, c0, false, false);
        c1 = __builtin_amdgcn_wmma_f32_16x16x32_f16(false, A1b, false, B1,
                                                    (short)0, c1, false, false);

        // dist[row, kcol] = e2 - 2*xe  (x2 row-constant: irrelevant to argmin)
#pragma unroll
        for (int i = 0; i < 8; ++i) {
            const float d0 = fmaf(-2.0f, c0[i], e2v);   // row = btile+hf*8+i
            if (d0 < bestv[i]) { bestv[i] = d0; bestk[i] = kcol; }
            const float d1 = fmaf(-2.0f, c1[i], e2v);   // row = +16
            if (d1 < bestv[8 + i]) { bestv[8 + i] = d1; bestk[8 + i] = kcol; }
        }
    }

    // Cross-lane argmin within each 16-lane half (xor 8,4,2,1 stays in half).
#pragma unroll
    for (int off = 8; off >= 1; off >>= 1) {
#pragma unroll
        for (int i = 0; i < 16; ++i) {
            const float ov = __shfl_xor(bestv[i], off, 32);
            const int   ok = __shfl_xor(bestk[i], off, 32);
            if (ov < bestv[i] || (ov == bestv[i] && ok < bestk[i])) {
                bestv[i] = ov; bestk[i] = ok;
            }
        }
    }
    if (l16 == 0) {
#pragma unroll
        for (int i = 0; i < 8; ++i) {
            kout[(btile + hf * 8 + i) * M_SUB + m]      = bestk[i];
            kout[(btile + 16 + hf * 8 + i) * M_SUB + m] = bestk[8 + i];
        }
    }
}

// ---------------------------------------------------------------------------
// Kernel 3: gather z_q (fp32 codebook) -> z_st, commit-loss partial sums,
// usage histogram. 16 (b,m) units per 256-thread block, 16 lanes per unit.
// z read & z_st write are single-touch streams -> non-temporal.
// ---------------------------------------------------------------------------
__global__ __launch_bounds__(256)
void gather_loss_kernel(const float* __restrict__ z,
                        const float* __restrict__ cb,
                        const int* __restrict__ kidx,
                        float* __restrict__ zst,
                        float* __restrict__ loss,
                        float* __restrict__ usage) {
    __shared__ float red[256];
    const int tid  = threadIdx.x;
    const int unit = blockIdx.x * 16 + (tid >> 4);   // (b,m) flat index
    const int sub  = tid & 15;
    const int b = unit >> 3;
    const int m = unit & 7;
    const int kk = kidx[unit];

    const v4f cv = *(const v4f*)(cb + (size_t)(m * K_CB + kk) * D_SUB + sub * 4);
    const size_t zo = (size_t)b * D_FULL + m * D_SUB + sub * 4;
    const v4f zv = __builtin_nontemporal_load((const v4f*)(z + zo));
    __builtin_nontemporal_store(cv, (v4f*)(zst + zo));   // z_st fwd == z_q

    const float dx = cv.x - zv.x, dy = cv.y - zv.y;
    const float dz = cv.z - zv.z, dw = cv.w - zv.w;
    red[tid] = dx * dx + dy * dy + dz * dz + dw * dw;

    if (sub == 0) atomicAdd(&usage[m * K_CB + kk], 1.0f);

    __syncthreads();
#pragma unroll
    for (int s = 128; s > 0; s >>= 1) {
        if (tid < s) red[tid] += red[tid + s];
        __syncthreads();
    }
    if (tid == 0) {
        // loss = BETA/M * (1/(B*d)) * total squared error
        const float scale = 0.25f / (8.0f * 16384.0f * 64.0f);
        atomicAdd(loss, red[0] * scale);
    }
}

// ---------------------------------------------------------------------------
// Kernel 4: perplexity per subspace m. One block per m.
// ---------------------------------------------------------------------------
__global__ __launch_bounds__(256)
void perplexity_kernel(const float* __restrict__ usage,
                       float* __restrict__ perp) {
    __shared__ float red[256];
    const int m = blockIdx.x;
    const int tid = threadIdx.x;
    float s = 0.0f;
    for (int kk = tid; kk < K_CB; kk += 256) {
        const float p = usage[m * K_CB + kk] * (1.0f / 16384.0f);
        s += p * logf(fmaxf(p, 1e-12f));
    }
    red[tid] = s;
    __syncthreads();
#pragma unroll
    for (int st = 128; st > 0; st >>= 1) {
        if (tid < st) red[tid] += red[tid + st];
        __syncthreads();
    }
    if (tid == 0) perp[m] = expf(-red[0]);   // exp(entropy), entropy = -sum
}

// ---------------------------------------------------------------------------
extern "C" void kernel_launch(void* const* d_in, const int* in_sizes, int n_in,
                              void* d_out, int out_size, void* d_ws, size_t ws_size,
                              hipStream_t stream) {
    const float* z  = (const float*)d_in[0];   // (16384, 512) fp32
    const float* cb = (const float*)d_in[1];   // (8, 1024, 64) fp32

    float* out  = (float*)d_out;
    float* zst  = out;                                        // B*D
    int*   kout = (int*)(out + (size_t)B_N * D_FULL);         // B*M (int32 bits)
    float* loss = out + (size_t)B_N * D_FULL + (size_t)B_N * M_SUB;  // scalar
    float* perp = loss + 1;                                   // (M,)

    char* ws = (char*)d_ws;
    half_t* cbh  = (half_t*)ws;                                            // 1 MB
    float*  e2   = (float*)(ws + (size_t)M_SUB * K_CB * D_SUB * sizeof(half_t)); // 32 KB
    float*  usage = e2 + M_SUB * K_CB;                                     // 32 KB

    (void)hipMemsetAsync(usage, 0, M_SUB * K_CB * sizeof(float), stream);
    (void)hipMemsetAsync(loss, 0, sizeof(float), stream);

    cvt_cb_e2_kernel<<<32, 256, 0, stream>>>(cb, cbh, e2);

    dim3 gridB(B_N / 256, M_SUB);   // 256 rows per block (8 waves x 32 rows)
    pq_argmin_kernel<<<gridB, 256, 0, stream>>>(z, cbh, e2, kout);

    gather_loss_kernel<<<(B_N * M_SUB) / 16, 256, 0, stream>>>(z, cb, kout, zst,
                                                               loss, usage);
    perplexity_kernel<<<M_SUB, 256, 0, stream>>>(usage, perp);
}
